// Qwen35GatedDeltaNet_14379550507565
// MI455X (gfx1250) — compile-verified, matrix operand order checked by reading, not run on hardware
//
#include <hip/hip_runtime.h>

// ---------------- problem dims (Qwen3.5 GatedDeltaNet) ----------------
#define Bb 2
#define Tt 2048
#define BT (Bb * Tt)          // 4096
#define HID 2048
#define HKh 16
#define HVh 32
#define DKd 128
#define DVd 128
#define KWW 4
#define KEY_DIM 2048
#define VAL_DIM 4096
#define CONV_DIM 8192

typedef __attribute__((ext_vector_type(16))) __bf16 v16bf;
typedef __attribute__((ext_vector_type(8)))  float  v8f;

// async-to-LDS builtin expects 'int __vector(4) *' (generic AS) per hipcc diagnostic
typedef int async_v4i __attribute__((vector_size(16)));
#define ASYNC_PTR(p) ((async_v4i*)(uintptr_t)(p))

#if __has_builtin(__builtin_amdgcn_global_load_async_to_lds_b128)
#define HAVE_ASYNC_LDS 1
#else
#define HAVE_ASYNC_LDS 0
#endif

// ---------------- f32 -> bf16 conversion (row-major copy) ----------------
__global__ void f32_to_bf16_kernel(const float* __restrict__ s, __bf16* __restrict__ d, size_t n) {
    size_t i = (size_t)blockIdx.x * blockDim.x + threadIdx.x;
    size_t stride = (size_t)gridDim.x * blockDim.x;
    for (; i < n; i += stride) d[i] = (__bf16)s[i];
}

// ---------------- f32 [K,N] -> bf16 [N,K] tiled transpose ----------------
__global__ __launch_bounds__(256) void f32_to_bf16_transpose_kernel(const float* __restrict__ s,
                                                                    __bf16* __restrict__ d,
                                                                    int rows, int cols) {
    __shared__ float tile[64][65];
    int rb = blockIdx.x * 64, cb = blockIdx.y * 64;
    int tx = threadIdx.x & 63, ty = threadIdx.x >> 6;   // 4 rows / iter
#pragma unroll
    for (int i = 0; i < 64; i += 4)
        tile[ty + i][tx] = s[(size_t)(rb + ty + i) * cols + cb + tx];
    __syncthreads();
#pragma unroll
    for (int i = 0; i < 64; i += 4)
        d[(size_t)(cb + ty + i) * rows + rb + tx] = (__bf16)tile[tx][ty + i];
}

// ---------------- WMMA bf16 GEMM: C[M,N] = A[M,K] @ Bt[N,K]^T, f32 accum ----------------
// block tile 128x128, 256 threads (8 waves), each wave 32x64 (2x4 wmma tiles), K-step 32.
// Double-buffered LDS tiles filled by gfx1250 async-to-LDS DMA, pipelined on ASYNCcnt:
// issue loads for tile k+1, then wait ASYNCcnt<=4 so tile k (issued last iter) is complete.
__global__ __launch_bounds__(256) void gemm_bf16_wmma(const __bf16* __restrict__ A,
                                                      const __bf16* __restrict__ Bt,
                                                      float* __restrict__ C,
                                                      int M, int N, int K) {
    __shared__ __align__(16) __bf16 As[2][128 * 32];   // [buf][m][k]
    __shared__ __align__(16) __bf16 Bs[2][128 * 32];   // [buf][n][k]

    const int tid  = threadIdx.x;
    const int lane = tid & 31;
    const int wave = tid >> 5;       // 0..7
    const int wm   = wave & 3;       // wave M slot: wm*32
    const int wn   = wave >> 2;      // wave N slot: wn*64
    const int lmod = lane & 15;
    const int lhi  = lane >> 4;      // 0/1
    const int bm   = blockIdx.x * 128;
    const int bn   = blockIdx.y * 128;

    const int srow = tid >> 1, shalf = tid & 1;     // staging: 128 rows x 2 halves
    const __bf16* a_g = A + (size_t)(bm + srow) * K + shalf * 16;
    const __bf16* b_g = Bt + (size_t)(bn + srow) * K + shalf * 16;
    const int soff = srow * 32 + shalf * 16;

    v8f acc[2][4];
#pragma unroll
    for (int i = 0; i < 2; ++i)
#pragma unroll
        for (int j = 0; j < 4; ++j)
#pragma unroll
            for (int r = 0; r < 8; ++r) acc[i][j][r] = 0.0f;

#if HAVE_ASYNC_LDS
    // prologue: stage K-tile 0 into buffer 0
    __builtin_amdgcn_global_load_async_to_lds_b128(ASYNC_PTR(a_g), ASYNC_PTR(&As[0][soff]), 0, 0);
    __builtin_amdgcn_global_load_async_to_lds_b128(ASYNC_PTR(a_g + 8), ASYNC_PTR(&As[0][soff + 8]), 0, 0);
    __builtin_amdgcn_global_load_async_to_lds_b128(ASYNC_PTR(b_g), ASYNC_PTR(&Bs[0][soff]), 0, 0);
    __builtin_amdgcn_global_load_async_to_lds_b128(ASYNC_PTR(b_g + 8), ASYNC_PTR(&Bs[0][soff + 8]), 0, 0);
#endif

    for (int k0 = 0; k0 < K; k0 += 32) {
        const int cur = (k0 >> 5) & 1;
        const int nxt = cur ^ 1;
#if HAVE_ASYNC_LDS
        if (k0 + 32 < K) {
            const int kn = k0 + 32;
            __builtin_amdgcn_global_load_async_to_lds_b128(ASYNC_PTR(a_g + kn), ASYNC_PTR(&As[nxt][soff]), 0, 0);
            __builtin_amdgcn_global_load_async_to_lds_b128(ASYNC_PTR(a_g + kn + 8), ASYNC_PTR(&As[nxt][soff + 8]), 0, 0);
            __builtin_amdgcn_global_load_async_to_lds_b128(ASYNC_PTR(b_g + kn), ASYNC_PTR(&Bs[nxt][soff]), 0, 0);
            __builtin_amdgcn_global_load_async_to_lds_b128(ASYNC_PTR(b_g + kn + 8), ASYNC_PTR(&Bs[nxt][soff + 8]), 0, 0);
            __builtin_amdgcn_s_wait_asynccnt(4);   // current tile's batch complete
        } else {
            __builtin_amdgcn_s_wait_asynccnt(0);
        }
#else
        // fallback: synchronous register staging into buffer `cur`
        uint4 a0 = ((const uint4*)(a_g + k0))[0];
        uint4 a1 = ((const uint4*)(a_g + k0))[1];
        uint4 b0 = ((const uint4*)(b_g + k0))[0];
        uint4 b1 = ((const uint4*)(b_g + k0))[1];
        ((uint4*)&As[cur][soff])[0] = a0;
        ((uint4*)&As[cur][soff])[1] = a1;
        ((uint4*)&Bs[cur][soff])[0] = b0;
        ((uint4*)&Bs[cur][soff])[1] = b1;
#endif
        if (k0 + 64 < K) {
            __builtin_prefetch(a_g + k0 + 64, 0, 1);
            __builtin_prefetch(b_g + k0 + 64, 0, 1);
        }
        __syncthreads();

        // fragments: lane covers row = tile*16 + lmod; elems 0..7 -> K = lhi*8 + e,
        // elems 8..15 -> K = 16 + lhi*8 + (e-8): two K-contiguous b128 LDS loads each.
        v16bf afrag[2];
#pragma unroll
        for (int i = 0; i < 2; ++i) {
            const __bf16* base = &As[cur][(wm * 32 + i * 16 + lmod) * 32];
            ((uint4*)&afrag[i])[0] = *(const uint4*)(base + lhi * 8);
            ((uint4*)&afrag[i])[1] = *(const uint4*)(base + 16 + lhi * 8);
        }
        v16bf bfrag[4];
#pragma unroll
        for (int j = 0; j < 4; ++j) {
            const __bf16* base = &Bs[cur][(wn * 64 + j * 16 + lmod) * 32];
            ((uint4*)&bfrag[j])[0] = *(const uint4*)(base + lhi * 8);
            ((uint4*)&bfrag[j])[1] = *(const uint4*)(base + 16 + lhi * 8);
        }
#pragma unroll
        for (int i = 0; i < 2; ++i)
#pragma unroll
            for (int j = 0; j < 4; ++j)
                acc[i][j] = __builtin_amdgcn_wmma_f32_16x16x32_bf16(
                    false, afrag[i], false, bfrag[j], (short)0, acc[i][j], false, false);
        __syncthreads();   // all reads of buf `cur` done -> safe to overwrite next iteration
    }

    // store: C/D layout — VGPR r: lanes0-15 M=r, lanes16-31 M=8+r; N=lmod
#pragma unroll
    for (int i = 0; i < 2; ++i)
#pragma unroll
        for (int j = 0; j < 4; ++j) {
            int col = bn + wn * 64 + j * 16 + lmod;
            int rowbase = bm + wm * 32 + i * 16 + lhi * 8;
#pragma unroll
            for (int r = 0; r < 8; ++r)
                C[(size_t)(rowbase + r) * N + col] = acc[i][j][r];
        }
}

// ---------------- beta / g projection: per (b,t) row ----------------
__global__ __launch_bounds__(256) void ba_proj_kernel(const float* __restrict__ x,
                                                      const float* __restrict__ Wb,
                                                      const float* __restrict__ Wa,
                                                      const float* __restrict__ dt_bias,
                                                      const float* __restrict__ A_log,
                                                      float* __restrict__ beta,
                                                      float* __restrict__ g) {
    int bt = blockIdx.x;
    int tid = threadIdx.x;
    __shared__ float xr[HID];
    for (int i = tid; i < HID; i += 256) xr[i] = x[(size_t)bt * HID + i];
    __syncthreads();
    int col = tid & 31, chunk = tid >> 5;   // 8 chunks of 256
    float pb = 0.f, pa = 0.f;
    int i0 = chunk * 256;
    for (int i = i0; i < i0 + 256; ++i) {
        float xv = xr[i];
        pb += xv * Wb[i * HVh + col];
        pa += xv * Wa[i * HVh + col];
    }
    __shared__ float rb[8][32], ra[8][32];
    rb[chunk][col] = pb;
    ra[chunk][col] = pa;
    __syncthreads();
    if (tid < HVh) {
        float sb = 0.f, sa = 0.f;
#pragma unroll
        for (int c = 0; c < 8; ++c) { sb += rb[c][tid]; sa += ra[c][tid]; }
        float be = 1.f / (1.f + __expf(-sb));
        float al = sa + dt_bias[tid];
        float sp = (al > 20.f) ? al : log1pf(__expf(al));
        beta[(size_t)bt * HVh + tid] = be;
        g[(size_t)bt * HVh + tid] = -__expf(A_log[tid]) * sp;
    }
}

// ---------------- fused reorder + causal depthwise conv(4) + SiLU + q/k L2 norm ----------------
__global__ __launch_bounds__(512) void conv_silu_norm_kernel(const float* __restrict__ qkv,
                                                             const float* __restrict__ conv_w,
                                                             float* __restrict__ qout,
                                                             float* __restrict__ kout,
                                                             float* __restrict__ vout) {
    int bt = blockIdx.x;
    int b = bt / Tt, t = bt % Tt;
    int tid = threadIdx.x;
    __shared__ float sums[32];           // 16 q-head + 16 k-head sum-of-squares
    if (tid < 32) sums[tid] = 0.f;
    __syncthreads();

    float vals[16];
    int c0 = tid * 16;                   // 16 contiguous channels, never cross head/region
    float lsq = 0.f;
#pragma unroll 4
    for (int j = 0; j < 16; ++j) {
        int c = c0 + j;
        int col;
        if (c < KEY_DIM)                  col = (c >> 7) * 512 + (c & 127);
        else if (c < 2 * KEY_DIM) { int c2 = c - KEY_DIM; col = (c2 >> 7) * 512 + 128 + (c2 & 127); }
        else                      { int c3 = c - 2 * KEY_DIM; col = (c3 >> 8) * 512 + 256 + (c3 & 255); }
        float acc = 0.f;
#pragma unroll
        for (int i = 0; i < KWW; ++i) {
            int tt = t - (KWW - 1) + i;
            float xv = (tt >= 0) ? qkv[(size_t)(b * Tt + tt) * CONV_DIM + col] : 0.f;
            acc += conv_w[c * KWW + i] * xv;
        }
        float sv = acc * (1.f / (1.f + __expf(-acc)));   // SiLU
        vals[j] = sv;
        if (c < 2 * KEY_DIM) lsq += sv * sv;
        else vout[(size_t)bt * VAL_DIM + (c - 2 * KEY_DIM)] = sv;
    }
    if (c0 < 2 * KEY_DIM) atomicAdd(&sums[c0 >> 7], lsq);
    __syncthreads();

    if (c0 < KEY_DIM) {
        float sc = rsqrtf(sums[c0 >> 7] + 1e-6f) * 0.08838834764831845f;  // * DK^-0.5
#pragma unroll
        for (int j = 0; j < 16; ++j) qout[(size_t)bt * KEY_DIM + c0 + j] = vals[j] * sc;
    } else if (c0 < 2 * KEY_DIM) {
        float sc = rsqrtf(sums[c0 >> 7] + 1e-6f);
#pragma unroll
        for (int j = 0; j < 16; ++j) kout[(size_t)bt * KEY_DIM + (c0 - KEY_DIM) + j] = vals[j] * sc;
    }
}

// ---------------- gated delta-rule recurrence ----------------
// one workgroup per (b, v-head); 512 threads; S[128][128] in registers, 32 per thread
__global__ __launch_bounds__(512) void recurrence_kernel(const float* __restrict__ q,
                                                         const float* __restrict__ k,
                                                         const float* __restrict__ v,
                                                         const float* __restrict__ g,
                                                         const float* __restrict__ beta,
                                                         float* __restrict__ o) {
    int bh = blockIdx.x;                 // 0..63
    int b = bh >> 5, h = bh & 31;
    int kh = h >> 1;                     // GQA repeat factor 2
    int tid = threadIdx.x;
    int vcol = tid & 127;
    int kq = tid >> 7;                   // k-slice 0..3 (32 rows each)

    float S[32];
#pragma unroll
    for (int i = 0; i < 32; ++i) S[i] = 0.f;

    __shared__ float lq[128], lk[128], lv[128], ldelta[128];
    __shared__ float part[4][128];
    __shared__ float sdecay, sbeta;

    for (int t = 0; t < Tt; ++t) {
        size_t bt = (size_t)b * Tt + t;
        size_t qoff = bt * KEY_DIM + (size_t)kh * DKd;
        size_t voff = bt * VAL_DIM + (size_t)h * DVd;
        if (tid < 128)      lq[tid]       = q[qoff + tid];
        else if (tid < 256) lk[tid - 128] = k[qoff + (tid - 128)];
        else if (tid < 384) lv[tid - 256] = v[voff + (tid - 256)];
        else if (tid == 384) {
            sdecay = __expf(g[bt * HVh + h]);
            sbeta  = beta[bt * HVh + h];
        }
        __syncthreads();

        float a = sdecay, bet = sbeta;
        float kvp = 0.f;
#pragma unroll
        for (int i = 0; i < 32; ++i) {
            S[i] *= a;
            kvp += lk[kq * 32 + i] * S[i];
        }
        part[kq][vcol] = kvp;
        __syncthreads();

        if (tid < 128) {
            float kv = part[0][tid] + part[1][tid] + part[2][tid] + part[3][tid];
            ldelta[tid] = bet * (lv[tid] - kv);
        }
        __syncthreads();

        float d = ldelta[vcol];
        float op = 0.f;
#pragma unroll
        for (int i = 0; i < 32; ++i) {
            S[i] += lk[kq * 32 + i] * d;
            op += lq[kq * 32 + i] * S[i];
        }
        part[kq][vcol] = op;
        __syncthreads();

        if (tid < 128)
            o[voff + tid] = part[0][tid] + part[1][tid] + part[2][tid] + part[3][tid];
        __syncthreads();
    }
}

// ---------------- gated RMSNorm -> bf16 ----------------
__global__ __launch_bounds__(128) void rmsnorm_gate_kernel(const float* __restrict__ o,
                                                           const float* __restrict__ z,
                                                           const float* __restrict__ w,
                                                           __bf16* __restrict__ y) {
    int row = blockIdx.x;                // 0 .. BT*HV-1
    int d = threadIdx.x;                 // 0..127
    size_t off = (size_t)row * DVd + d;
    float x = o[off];
    float s = x * x;
#pragma unroll
    for (int m = 16; m > 0; m >>= 1) s += __shfl_xor(s, m, 32);
    __shared__ float red[4];
    if ((d & 31) == 0) red[d >> 5] = s;
    __syncthreads();
    float var = (red[0] + red[1] + red[2] + red[3]) * (1.0f / DVd);
    float xn = x * rsqrtf(var + 1e-6f);
    float zz = z[off];
    float gate = zz * (1.f / (1.f + __expf(-zz)));
    y[off] = (__bf16)(xn * gate * w[d]);
}

// ---------------- host launcher ----------------
extern "C" void kernel_launch(void* const* d_in, const int* in_sizes, int n_in,
                              void* d_out, int out_size, void* d_ws, size_t ws_size,
                              hipStream_t stream) {
    const float* x        = (const float*)d_in[0];
    const float* W_qkv    = (const float*)d_in[1];
    const float* W_z      = (const float*)d_in[2];
    const float* W_b      = (const float*)d_in[3];
    const float* W_a      = (const float*)d_in[4];
    const float* conv_w   = (const float*)d_in[5];
    const float* dt_bias  = (const float*)d_in[6];
    const float* A_log    = (const float*)d_in[7];
    const float* norm_w   = (const float*)d_in[8];
    const float* W_out    = (const float*)d_in[9];
    float* out = (float*)d_out;

    char* ws = (char*)d_ws;
    constexpr size_t SZ_XBF   = (size_t)BT * HID * 2;
    constexpr size_t SZ_WQKV  = (size_t)HID * CONV_DIM * 2;
    constexpr size_t SZ_WZ    = (size_t)HID * VAL_DIM * 2;
    constexpr size_t SZ_WOUT  = (size_t)VAL_DIM * HID * 2;
    constexpr size_t SZ_QKV   = (size_t)BT * CONV_DIM * 4;     // reused for o + y_bf16
    constexpr size_t SZ_Q     = (size_t)BT * KEY_DIM * 4;
    constexpr size_t SZ_K     = (size_t)BT * KEY_DIM * 4;
    constexpr size_t SZ_V     = (size_t)BT * VAL_DIM * 4;
    constexpr size_t SZ_Z     = (size_t)BT * VAL_DIM * 4;
    constexpr size_t SZ_BG    = (size_t)BT * HVh * 4;

    size_t off = 0;
    __bf16* x_bf    = (__bf16*)(ws + off); off += SZ_XBF;
    __bf16* wqkv_bt = (__bf16*)(ws + off); off += SZ_WQKV;   // [CONV_DIM, HID]
    __bf16* wz_bt   = (__bf16*)(ws + off); off += SZ_WZ;     // [VAL_DIM, HID]
    __bf16* wout_bt = (__bf16*)(ws + off); off += SZ_WOUT;   // [HID, VAL_DIM]
    float*  qkv     = (float*)(ws + off);  size_t qkv_off = off; off += SZ_QKV;
    float*  qbuf    = (float*)(ws + off);  off += SZ_Q;
    float*  kbuf    = (float*)(ws + off);  off += SZ_K;
    float*  vbuf    = (float*)(ws + off);  off += SZ_V;
    float*  zbuf    = (float*)(ws + off);  off += SZ_Z;
    float*  betab   = (float*)(ws + off);  off += SZ_BG;
    float*  gbuf    = (float*)(ws + off);  off += SZ_BG;
    float*  obuf = (float*)(ws + qkv_off);
    __bf16* y_bf = (__bf16*)(ws + qkv_off + (size_t)BT * VAL_DIM * 4);

    // 1. staging: x row-major bf16; weights transposed to [N,K] bf16
    f32_to_bf16_kernel<<<2048, 256, 0, stream>>>(x, x_bf, (size_t)BT * HID);
    f32_to_bf16_transpose_kernel<<<dim3(HID / 64, CONV_DIM / 64), 256, 0, stream>>>(
        W_qkv, wqkv_bt, HID, CONV_DIM);
    f32_to_bf16_transpose_kernel<<<dim3(HID / 64, VAL_DIM / 64), 256, 0, stream>>>(
        W_z, wz_bt, HID, VAL_DIM);
    f32_to_bf16_transpose_kernel<<<dim3(VAL_DIM / 64, HID / 64), 256, 0, stream>>>(
        W_out, wout_bt, VAL_DIM, HID);

    // 2. qkv = x @ W_qkv
    gemm_bf16_wmma<<<dim3(BT / 128, CONV_DIM / 128), 256, 0, stream>>>(
        x_bf, wqkv_bt, qkv, BT, CONV_DIM, HID);
    // 3. z = x @ W_z
    gemm_bf16_wmma<<<dim3(BT / 128, VAL_DIM / 128), 256, 0, stream>>>(
        x_bf, wz_bt, zbuf, BT, VAL_DIM, HID);
    // 4. beta / g
    ba_proj_kernel<<<BT, 256, 0, stream>>>(x, W_b, W_a, dt_bias, A_log, betab, gbuf);
    // 5. reorder + conv + SiLU + q/k L2norm
    conv_silu_norm_kernel<<<BT, 512, 0, stream>>>(qkv, conv_w, qbuf, kbuf, vbuf);
    // 6. gated delta-rule scan (state in registers)
    recurrence_kernel<<<Bb * HVh, 512, 0, stream>>>(qbuf, kbuf, vbuf, gbuf, betab, obuf);
    // 7. gated RMSNorm -> bf16
    rmsnorm_gate_kernel<<<BT * HVh, 128, 0, stream>>>(obuf, zbuf, norm_w, y_bf);
    // 8. out = y @ W_out
    gemm_bf16_wmma<<<dim3(BT / 128, HID / 128), 256, 0, stream>>>(
        y_bf, wout_bt, out, BT, HID, VAL_DIM);
}
